// SelfAttention_62440234549817
// MI455X (gfx1250) — compile-verified, matrix operand order checked by reading, not run on hardware
//
#include <hip/hip_runtime.h>
#include <stdint.h>

// ---------------------------------------------------------------------------
// MI455X (gfx1250) causal multi-head self-attention, bf16 WMMA path.
// D=1024, H=16, d_head=64, B=2, T=2048.  All matmuls via
// v_wmma_f32_16x16x32_bf16 (fp32 accumulate).
// ---------------------------------------------------------------------------

typedef __attribute__((ext_vector_type(16))) __bf16 v16bf;
typedef __attribute__((ext_vector_type(8)))  float  v8f;

union BF16Frag {
    v16bf v;
    uint4 q[2];
};

static constexpr int TSEQ  = 2048;
static constexpr int DM    = 1024;
static constexpr int NH    = 16;
static constexpr int DH    = 64;
static constexpr int BATCH = 2;
static constexpr int MTOT  = BATCH * TSEQ;   // 4096

__device__ __forceinline__ unsigned short f2bf(float f) {
    unsigned u = __float_as_uint(f);
    u += 0x7FFFu + ((u >> 16) & 1u);          // round-to-nearest-even
    return (unsigned short)(u >> 16);
}

// --------------------------- conversion kernels ----------------------------

__global__ void cvt_bf16(const float* __restrict__ in,
                         unsigned short* __restrict__ out, int n) {
    int i = (blockIdx.x * blockDim.x + threadIdx.x) * 4;
    if (i + 3 < n) {
        float4 f = *(const float4*)(in + i);
        uint2 p;
        p.x = (unsigned)f2bf(f.x) | ((unsigned)f2bf(f.y) << 16);
        p.y = (unsigned)f2bf(f.z) | ((unsigned)f2bf(f.w) << 16);
        *(uint2*)(out + i) = p;
    }
}

// W[k][n] fp32 -> Wt[n][k] bf16 (1024x1024)
__global__ void cvt_bf16_T(const float* __restrict__ in,
                           unsigned short* __restrict__ out) {
    int idx = blockIdx.x * blockDim.x + threadIdx.x;   // over 1024*1024
    int k = idx >> 10, n = idx & 1023;
    out[(size_t)n * DM + k] = f2bf(in[idx]);
}

// ------------------------------ QKV GEMM -----------------------------------
// C[4096 x 1024] = Xbf[4096 x 1024] * W (W given transposed, row = N).
// Block tile 128x128, 8 waves, each wave 32(M) x 64(N) = 2x4 WMMA accums.

__global__ __launch_bounds__(256)
void gemm_qkv(const unsigned short* __restrict__ X,
              const unsigned short* __restrict__ Wqt,
              const unsigned short* __restrict__ Wkt,
              const unsigned short* __restrict__ Wvt,
              unsigned short* __restrict__ Qb,     // [B,H,T,64], pre-scaled
              unsigned short* __restrict__ Kb,     // [B,H,T,64]
              unsigned short* __restrict__ Vt)     // [B,H,64,T]
{
    constexpr int BM = 128, BN = 128, BK = 32, LS = 48;  // LS*2B = 96B rows
    __shared__ unsigned short As[BM * LS];
    __shared__ unsigned short Bs[BN * LS];

    const int proj = blockIdx.z;
    const unsigned short* Bt = (proj == 0) ? Wqt : (proj == 1) ? Wkt : Wvt;

    const int m0 = blockIdx.x * BM, n0 = blockIdx.y * BN;
    const int tid = threadIdx.x;
    const int lane = tid & 31, wid = tid >> 5;
    const int lrow = lane & 15, lhalf = lane >> 4, kb = lhalf * 8;
    const int wm = wid & 3, wn = wid >> 2;

    v8f zero = {};
    v8f acc[2][4];
#pragma unroll
    for (int i = 0; i < 2; ++i)
#pragma unroll
        for (int j = 0; j < 4; ++j) acc[i][j] = zero;

    const int ldr = tid >> 1;                 // 0..127 tile row
    const int ldc = (tid & 1) * 16;           // 0 or 16

    for (int k0 = 0; k0 < DM; k0 += BK) {
        uint4 a0 = *(const uint4*)(X  + (size_t)(m0 + ldr) * DM + k0 + ldc);
        uint4 a1 = *(const uint4*)(X  + (size_t)(m0 + ldr) * DM + k0 + ldc + 8);
        uint4 b0 = *(const uint4*)(Bt + (size_t)(n0 + ldr) * DM + k0 + ldc);
        uint4 b1 = *(const uint4*)(Bt + (size_t)(n0 + ldr) * DM + k0 + ldc + 8);
        __syncthreads();                       // prior compute done
        *(uint4*)(As + ldr * LS + ldc)     = a0;
        *(uint4*)(As + ldr * LS + ldc + 8) = a1;
        *(uint4*)(Bs + ldr * LS + ldc)     = b0;
        *(uint4*)(Bs + ldr * LS + ldc + 8) = b1;
        __syncthreads();

        BF16Frag af[2], bf[4];
#pragma unroll
        for (int mi = 0; mi < 2; ++mi) {
            const unsigned short* p = As + (wm * 32 + mi * 16 + lrow) * LS + kb;
            af[mi].q[0] = *(const uint4*)(p);
            af[mi].q[1] = *(const uint4*)(p + 16);
        }
#pragma unroll
        for (int ni = 0; ni < 4; ++ni) {
            const unsigned short* p = Bs + (wn * 64 + ni * 16 + lrow) * LS + kb;
            bf[ni].q[0] = *(const uint4*)(p);
            bf[ni].q[1] = *(const uint4*)(p + 16);
        }
#pragma unroll
        for (int mi = 0; mi < 2; ++mi)
#pragma unroll
            for (int ni = 0; ni < 4; ++ni)
                acc[mi][ni] = __builtin_amdgcn_wmma_f32_16x16x32_bf16(
                    false, af[mi].v, false, bf[ni].v,
                    (short)0, acc[mi][ni], false, false);
    }

    // scatter epilogue into head-major layouts
#pragma unroll
    for (int mi = 0; mi < 2; ++mi) {
#pragma unroll
        for (int ni = 0; ni < 4; ++ni) {
            int gn = n0 + wn * 64 + ni * 16 + lrow;
            int hh = gn >> 6, dd = gn & 63;
#pragma unroll
            for (int r = 0; r < 8; ++r) {
                int gm = m0 + wm * 32 + mi * 16 + lhalf * 8 + r;
                int bb = gm >> 11, tt = gm & 2047;
                float va = acc[mi][ni][r];
                if (proj == 0) {
                    Qb[(((size_t)bb * NH + hh) * TSEQ + tt) * DH + dd] =
                        f2bf(va * 0.125f);     // 1/sqrt(64)
                } else if (proj == 1) {
                    Kb[(((size_t)bb * NH + hh) * TSEQ + tt) * DH + dd] = f2bf(va);
                } else {
                    Vt[(((size_t)bb * NH + hh) * DH + dd) * TSEQ + tt] = f2bf(va);
                }
            }
        }
    }
}

// --------------------------- flash attention -------------------------------
// One wave = one 16-row query tile of one (b,h). Streams 64-wide key tiles.

__global__ __launch_bounds__(256)
void attn_fwd(const unsigned short* __restrict__ Qb,
              const unsigned short* __restrict__ Kb,
              const unsigned short* __restrict__ Vt,
              unsigned short* __restrict__ Z)      // [B,T,1024] bf16
{
    constexpr int PLS = 72;                        // 144B rows, 16B aligned
    __shared__ unsigned short Plds[8 * 16 * PLS];

    const int tid = threadIdx.x, lane = tid & 31, w = tid >> 5;
    const int lrow = lane & 15, lhalf = lane >> 4, kb = lhalf * 8;

    int gtile = blockIdx.x * 8 + w;                // 4096 tiles total
    int qt = gtile & 127;                          // 128 q-tiles per head
    int bh = gtile >> 7;
    int hh = bh & (NH - 1), bb = bh >> 4;
    int qbase = qt * 16;

    const size_t head = ((size_t)bb * NH + hh) * (size_t)TSEQ * DH;
    const unsigned short* Qh = Qb + head;
    const unsigned short* Kh = Kb + head;
    const unsigned short* Vh = Vt + head;          // [64][T]

    BF16Frag qf[2];
#pragma unroll
    for (int ks = 0; ks < 2; ++ks) {
        const unsigned short* p = Qh + (size_t)(qbase + lrow) * DH + ks * 32 + kb;
        qf[ks].q[0] = *(const uint4*)p;
        qf[ks].q[1] = *(const uint4*)(p + 16);
    }

    v8f zero = {};
    v8f o[4];
#pragma unroll
    for (int nd = 0; nd < 4; ++nd) o[nd] = zero;
    float mrow[8], lsum[8];
#pragma unroll
    for (int r = 0; r < 8; ++r) { mrow[r] = -__builtin_inff(); lsum[r] = 0.f; }

    unsigned short* Pw = Plds + w * 16 * PLS;      // wave-private
    const int nkt = (qbase + 16 + 63) >> 6;        // causal: tiles up to diag

    for (int kt = 0; kt < nkt; ++kt) {
        const int ktb = kt * 64;

        // S = Q * K^T  (16 x 64), 4 N-subtiles x 2 K-steps
        v8f s[4];
#pragma unroll
        for (int ni = 0; ni < 4; ++ni) {
            s[ni] = zero;
#pragma unroll
            for (int ks = 0; ks < 2; ++ks) {
                BF16Frag kf;
                const unsigned short* p =
                    Kh + (size_t)(ktb + ni * 16 + lrow) * DH + ks * 32 + kb;
                kf.q[0] = *(const uint4*)p;
                kf.q[1] = *(const uint4*)(p + 16);
                s[ni] = __builtin_amdgcn_wmma_f32_16x16x32_bf16(
                    false, qf[ks].v, false, kf.v, (short)0, s[ni], false, false);
            }
        }

        // causal mask (C-layout: row = qbase+lhalf*8+r, col = ktb+ni*16+lrow)
#pragma unroll
        for (int ni = 0; ni < 4; ++ni) {
            int col = ktb + ni * 16 + lrow;
#pragma unroll
            for (int r = 0; r < 8; ++r) {
                int row = qbase + lhalf * 8 + r;
                if (col > row) s[ni][r] = -__builtin_inff();
            }
        }

        // online softmax per row (reduce across 16-lane halves)
        float scale_r[8];
#pragma unroll
        for (int r = 0; r < 8; ++r) {
            float v = fmaxf(fmaxf(s[0][r], s[1][r]), fmaxf(s[2][r], s[3][r]));
#pragma unroll
            for (int m = 1; m < 16; m <<= 1) v = fmaxf(v, __shfl_xor(v, m, 32));
            float mnew = fmaxf(mrow[r], v);
            float sc = __expf(mrow[r] - mnew);
            float psum = 0.f;
#pragma unroll
            for (int ni = 0; ni < 4; ++ni) {
                float p = __expf(s[ni][r] - mnew);
                psum += p;
                Pw[(lhalf * 8 + r) * PLS + ni * 16 + lrow] = f2bf(p);
            }
#pragma unroll
            for (int m = 1; m < 16; m <<= 1) psum += __shfl_xor(psum, m, 32);
            lsum[r] = lsum[r] * sc + psum;
            mrow[r] = mnew;
            scale_r[r] = sc;
        }
#pragma unroll
        for (int nd = 0; nd < 4; ++nd)
#pragma unroll
            for (int r = 0; r < 8; ++r) o[nd][r] *= scale_r[r];

        // wave-local cross-lane LDS dependency: wait for DS stores
        asm volatile("s_wait_dscnt 0" ::: "memory");

        // P (A-layout) from LDS, V (B-layout) from transposed global V
        BF16Frag pf[2];
#pragma unroll
        for (int ks = 0; ks < 2; ++ks) {
            const unsigned short* p = Pw + lrow * PLS + ks * 32 + kb;
            pf[ks].q[0] = *(const uint4*)p;
            pf[ks].q[1] = *(const uint4*)(p + 16);
        }
#pragma unroll
        for (int nd = 0; nd < 4; ++nd) {
#pragma unroll
            for (int ks = 0; ks < 2; ++ks) {
                BF16Frag vf;
                const unsigned short* p =
                    Vh + (size_t)(nd * 16 + lrow) * TSEQ + ktb + ks * 32 + kb;
                vf.q[0] = *(const uint4*)p;
                vf.q[1] = *(const uint4*)(p + 16);
                o[nd] = __builtin_amdgcn_wmma_f32_16x16x32_bf16(
                    false, pf[ks].v, false, vf.v, (short)0, o[nd], false, false);
            }
        }
    }

    // normalize + store Z as [B, T, H*64] bf16
#pragma unroll
    for (int nd = 0; nd < 4; ++nd) {
#pragma unroll
        for (int r = 0; r < 8; ++r) {
            int row = qbase + lhalf * 8 + r;
            float v = o[nd][r] / lsum[r];
            Z[((size_t)bb * TSEQ + row) * DM + hh * DH + nd * 16 + lrow] = f2bf(v);
        }
    }
}

// --------------------------- output projection -----------------------------

__global__ __launch_bounds__(256)
void gemm_out(const unsigned short* __restrict__ Zb,
              const unsigned short* __restrict__ Wot,   // [N][K] bf16
              float* __restrict__ Out)                  // [4096][1024] fp32
{
    constexpr int BM = 128, BN = 128, BK = 32, LS = 48;
    __shared__ unsigned short As[BM * LS];
    __shared__ unsigned short Bs[BN * LS];

    const int m0 = blockIdx.x * BM, n0 = blockIdx.y * BN;
    const int tid = threadIdx.x;
    const int lane = tid & 31, wid = tid >> 5;
    const int lrow = lane & 15, lhalf = lane >> 4, kb = lhalf * 8;
    const int wm = wid & 3, wn = wid >> 2;

    v8f zero = {};
    v8f acc[2][4];
#pragma unroll
    for (int i = 0; i < 2; ++i)
#pragma unroll
        for (int j = 0; j < 4; ++j) acc[i][j] = zero;

    const int ldr = tid >> 1;
    const int ldc = (tid & 1) * 16;

    for (int k0 = 0; k0 < DM; k0 += BK) {
        uint4 a0 = *(const uint4*)(Zb  + (size_t)(m0 + ldr) * DM + k0 + ldc);
        uint4 a1 = *(const uint4*)(Zb  + (size_t)(m0 + ldr) * DM + k0 + ldc + 8);
        uint4 b0 = *(const uint4*)(Wot + (size_t)(n0 + ldr) * DM + k0 + ldc);
        uint4 b1 = *(const uint4*)(Wot + (size_t)(n0 + ldr) * DM + k0 + ldc + 8);
        __syncthreads();
        *(uint4*)(As + ldr * LS + ldc)     = a0;
        *(uint4*)(As + ldr * LS + ldc + 8) = a1;
        *(uint4*)(Bs + ldr * LS + ldc)     = b0;
        *(uint4*)(Bs + ldr * LS + ldc + 8) = b1;
        __syncthreads();

        BF16Frag af[2], bf[4];
#pragma unroll
        for (int mi = 0; mi < 2; ++mi) {
            const unsigned short* p = As + (wm * 32 + mi * 16 + lrow) * LS + kb;
            af[mi].q[0] = *(const uint4*)(p);
            af[mi].q[1] = *(const uint4*)(p + 16);
        }
#pragma unroll
        for (int ni = 0; ni < 4; ++ni) {
            const unsigned short* p = Bs + (wn * 64 + ni * 16 + lrow) * LS + kb;
            bf[ni].q[0] = *(const uint4*)(p);
            bf[ni].q[1] = *(const uint4*)(p + 16);
        }
#pragma unroll
        for (int mi = 0; mi < 2; ++mi)
#pragma unroll
            for (int ni = 0; ni < 4; ++ni)
                acc[mi][ni] = __builtin_amdgcn_wmma_f32_16x16x32_bf16(
                    false, af[mi].v, false, bf[ni].v,
                    (short)0, acc[mi][ni], false, false);
    }

#pragma unroll
    for (int mi = 0; mi < 2; ++mi)
#pragma unroll
        for (int ni = 0; ni < 4; ++ni) {
            int gn = n0 + wn * 64 + ni * 16 + lrow;
#pragma unroll
            for (int r = 0; r < 8; ++r) {
                int gm = m0 + wm * 32 + mi * 16 + lhalf * 8 + r;
                Out[(size_t)gm * DM + gn] = acc[mi][ni][r];
            }
        }
}

// ------------------------------- launcher ----------------------------------

extern "C" void kernel_launch(void* const* d_in, const int* in_sizes, int n_in,
                              void* d_out, int out_size, void* d_ws, size_t ws_size,
                              hipStream_t stream) {
    const float* x  = (const float*)d_in[0];
    const float* Wq = (const float*)d_in[1];
    const float* Wk = (const float*)d_in[2];
    const float* Wv = (const float*)d_in[3];
    const float* Wo = (const float*)d_in[4];
    float* out = (float*)d_out;

    // workspace layout (bf16 elements), total 24M elems = 48 MB
    unsigned short* Xb  = (unsigned short*)d_ws;               // 4096*1024
    unsigned short* Wqt = Xb  + (size_t)MTOT * DM;
    unsigned short* Wkt = Wqt + (size_t)DM * DM;
    unsigned short* Wvt = Wkt + (size_t)DM * DM;
    unsigned short* Wot = Wvt + (size_t)DM * DM;
    unsigned short* Qb  = Wot + (size_t)DM * DM;
    unsigned short* Kb  = Qb  + (size_t)BATCH * NH * TSEQ * DH;
    unsigned short* Vt  = Kb  + (size_t)BATCH * NH * TSEQ * DH;
    unsigned short* Zb  = Vt  + (size_t)BATCH * NH * TSEQ * DH;

    cvt_bf16<<<4096, 256, 0, stream>>>(x, Xb, MTOT * DM);
    cvt_bf16_T<<<4096, 256, 0, stream>>>(Wq, Wqt);
    cvt_bf16_T<<<4096, 256, 0, stream>>>(Wk, Wkt);
    cvt_bf16_T<<<4096, 256, 0, stream>>>(Wv, Wvt);
    cvt_bf16_T<<<4096, 256, 0, stream>>>(Wo, Wot);

    dim3 gqkv(MTOT / 128, DM / 128, 3);          // 32 x 8 x 3
    gemm_qkv<<<gqkv, 256, 0, stream>>>(Xb, Wqt, Wkt, Wvt, Qb, Kb, Vt);

    attn_fwd<<<(BATCH * NH * (TSEQ / 16)) / 8, 256, 0, stream>>>(Qb, Kb, Vt, Zb);

    dim3 gout(MTOT / 128, DM / 128);             // 32 x 8
    gemm_out<<<gout, 256, 0, stream>>>(Zb, Wot, out);
}
